// GnnMapping_26173530702019
// MI455X (gfx1250) — matrix operand order repaired and checked by reading, not compiled, not run on hardware
//
#include <hip/hip_runtime.h>

// ---------------- problem constants ----------------
#define B_    128
#define Q_    32
#define F_    64
#define NGATE 256
#define NC_   (B_*NGATE)   // 32768 circuit nodes
#define NT_   (B_*Q_)      // 4096 topo nodes
#define EC_   131072       // circuit edges (before self loops)
#define ET_   16384        // topo edges (before self loops)
#define NPAIR (B_*Q_*Q_)   // 131072

typedef __attribute__((ext_vector_type(16))) _Float16 v16h;
typedef __attribute__((ext_vector_type(8)))  _Float16 v8h;
typedef __attribute__((ext_vector_type(8)))  float    v8f;

__device__ __forceinline__ float leakyf(float x, float s) { return x > 0.f ? x : s * x; }

__device__ __forceinline__ void atomicMaxF(float* addr, float val) {
  if (val >= 0.f) atomicMax((int*)addr, __float_as_int(val));
  else            atomicMin((unsigned int*)addr, (unsigned int)__float_as_int(val));
}

// ---------------- weight convert: W[Kin,Co] fp32 (rows r0..r0+Kin) -> Wt[Co,Kin] f16 ----
__global__ void k_convert(const float* __restrict__ W, _Float16* __restrict__ Wt,
                          int Kin, int Co, int r0) {
  int gid = blockIdx.x * blockDim.x + threadIdx.x;
  if (gid >= Kin * Co) return;
  int o = gid / Kin, k = gid - o * Kin;
  Wt[(size_t)o * Kin + k] = (_Float16)W[(size_t)(r0 + k) * Co + o];
}

// ---------------- activation convert: X[N*Ci] fp32 -> Xh f16 --------------------------
__global__ void k_cvt_act(const float* __restrict__ X, _Float16* __restrict__ Xh, int n) {
  int gid = blockIdx.x * blockDim.x + threadIdx.x;
  if (gid >= n) return;
  Xh[gid] = (_Float16)X[gid];
}

// ---------------- WMMA GEMM: Y[N,Co] = Xh[N,Ci] @ W + bias (optional leaky 0.01) ------
// Xh f16 row-major; Wt f16 transposed [Co,Ci]. One wave computes a 16x64 tile
// (4 accumulators); K stepped by 32: 2 A-loads feed 4 WMMAs.
__global__ __launch_bounds__(256)
void k_gemm(const _Float16* __restrict__ Xh, const _Float16* __restrict__ Wt,
            const float* __restrict__ bias, float* __restrict__ Y,
            int nrows, int Ci, int Co, int act) {
  int lane = threadIdx.x & 31;
  int wid  = blockIdx.x * (blockDim.x >> 5) + (threadIdx.x >> 5);
  int ntn  = Co >> 6;                       // 64-wide column tiles
  int total = (nrows >> 4) * ntn;
  if (wid >= total) return;
  int tn = wid % ntn, tm = wid / ntn;
  int l15 = lane & 15, h = lane >> 4;
  int rowA  = (tm << 4) + l15;
  int colB0 = (tn << 6) + l15;
  const _Float16* xrow = Xh + (size_t)rowA * Ci;
  const _Float16* w0 = Wt + (size_t)(colB0)      * Ci;
  const _Float16* w1 = Wt + (size_t)(colB0 + 16) * Ci;
  const _Float16* w2 = Wt + (size_t)(colB0 + 32) * Ci;
  const _Float16* w3 = Wt + (size_t)(colB0 + 48) * Ci;
  v8f acc0 = {}, acc1 = {}, acc2 = {}, acc3 = {};
  for (int k0 = 0; k0 < Ci; k0 += 32) {
    v8h alo = *(const v8h*)(xrow + k0 + 16 * h);
    v8h ahi = *(const v8h*)(xrow + k0 + 16 * h + 8);
    v16h a;
#pragma unroll
    for (int e = 0; e < 8; ++e) { a[e] = alo[e]; a[8 + e] = ahi[e]; }
    v16h b0 = *(const v16h*)(w0 + k0 + 16 * h);
    v16h b1 = *(const v16h*)(w1 + k0 + 16 * h);
    v16h b2 = *(const v16h*)(w2 + k0 + 16 * h);
    v16h b3 = *(const v16h*)(w3 + k0 + 16 * h);
    acc0 = __builtin_amdgcn_wmma_f32_16x16x32_f16(false, a, false, b0, (short)0, acc0, false, false);
    acc1 = __builtin_amdgcn_wmma_f32_16x16x32_f16(false, a, false, b1, (short)0, acc1, false, false);
    acc2 = __builtin_amdgcn_wmma_f32_16x16x32_f16(false, a, false, b2, (short)0, acc2, false, false);
    acc3 = __builtin_amdgcn_wmma_f32_16x16x32_f16(false, a, false, b3, (short)0, acc3, false, false);
  }
#pragma unroll
  for (int j = 0; j < 4; ++j) {
    v8f acc = j == 0 ? acc0 : (j == 1 ? acc1 : (j == 2 ? acc2 : acc3));
    int colB = colB0 + 16 * j;
    float bs = bias ? bias[colB] : 0.f;
#pragma unroll
    for (int v = 0; v < 8; ++v) {
      int orow = (tm << 4) + v + 8 * h;
      float val = acc[v] + bs;
      if (act) val = leakyf(val, 0.01f);
      Y[(size_t)orow * Co + colB] = val;
    }
  }
}

// ---------------- fused pair GEMM: h2 = leaky( leaky(U_i + V_j) @ W2 + b2 ) -----------
// rows r = b*1024 + i*32 + j; A[r,k] = leaky01(U[(b,i),k] + V[(b,j),k]); Ci=384, Co=128.
// Wave computes 16x64 (4 accumulators) so the A synthesis+cvt feeds 4 WMMAs.
__global__ __launch_bounds__(256)
void k_gemm_pair(const float* __restrict__ U, const float* __restrict__ V,
                 const _Float16* __restrict__ Wt, const float* __restrict__ bias,
                 float* __restrict__ Y) {
  const int Ci = 6 * F_;   // 384
  const int Co = 2 * F_;   // 128
  int lane = threadIdx.x & 31;
  int wid  = blockIdx.x * 8 + (threadIdx.x >> 5);
  int ntn  = Co >> 6;                        // 2
  int total = (NPAIR >> 4) * ntn;            // 16384
  if (wid >= total) return;
  int tn = wid % ntn, tm = wid / ntn;
  int l15 = lane & 15, h = lane >> 4;
  int rowA = (tm << 4) + l15;
  int bI = rowA >> 10, iI = (rowA >> 5) & 31, jI = rowA & 31;
  const float* urow = U + (size_t)(bI * Q_ + iI) * Ci;
  const float* vrow = V + (size_t)(bI * Q_ + jI) * Ci;
  int colB0 = (tn << 6) + l15;
  const _Float16* w0 = Wt + (size_t)(colB0)      * Ci;
  const _Float16* w1 = Wt + (size_t)(colB0 + 16) * Ci;
  const _Float16* w2 = Wt + (size_t)(colB0 + 32) * Ci;
  const _Float16* w3 = Wt + (size_t)(colB0 + 48) * Ci;
  v8f acc0 = {}, acc1 = {}, acc2 = {}, acc3 = {};
  for (int k0 = 0; k0 < Ci; k0 += 32) {
    const float4* u0 = (const float4*)(urow + k0 + 16 * h);
    const float4* u2 = (const float4*)(urow + k0 + 16 * h + 8);
    const float4* p0 = (const float4*)(vrow + k0 + 16 * h);
    const float4* p2 = (const float4*)(vrow + k0 + 16 * h + 8);
    float4 a0 = u0[0], a1 = u0[1], a2 = u2[0], a3 = u2[1];
    float4 c0 = p0[0], c1 = p0[1], c2 = p2[0], c3 = p2[1];
    float af[16] = {a0.x+c0.x,a0.y+c0.y,a0.z+c0.z,a0.w+c0.w,
                    a1.x+c1.x,a1.y+c1.y,a1.z+c1.z,a1.w+c1.w,
                    a2.x+c2.x,a2.y+c2.y,a2.z+c2.z,a2.w+c2.w,
                    a3.x+c3.x,a3.y+c3.y,a3.z+c3.z,a3.w+c3.w};
    v16h a;
#pragma unroll
    for (int e = 0; e < 16; ++e) a[e] = (_Float16)leakyf(af[e], 0.01f);
    v16h b0 = *(const v16h*)(w0 + k0 + 16 * h);
    v16h b1 = *(const v16h*)(w1 + k0 + 16 * h);
    v16h b2 = *(const v16h*)(w2 + k0 + 16 * h);
    v16h b3 = *(const v16h*)(w3 + k0 + 16 * h);
    acc0 = __builtin_amdgcn_wmma_f32_16x16x32_f16(false, a, false, b0, (short)0, acc0, false, false);
    acc1 = __builtin_amdgcn_wmma_f32_16x16x32_f16(false, a, false, b1, (short)0, acc1, false, false);
    acc2 = __builtin_amdgcn_wmma_f32_16x16x32_f16(false, a, false, b2, (short)0, acc2, false, false);
    acc3 = __builtin_amdgcn_wmma_f32_16x16x32_f16(false, a, false, b3, (short)0, acc3, false, false);
  }
#pragma unroll
  for (int j = 0; j < 4; ++j) {
    v8f acc = j == 0 ? acc0 : (j == 1 ? acc1 : (j == 2 ? acc2 : acc3));
    int colB = colB0 + 16 * j;
    float bs = bias[colB];
#pragma unroll
    for (int v = 0; v < 8; ++v) {
      int orow = (tm << 4) + v + 8 * h;
      Y[(size_t)orow * Co + colB] = leakyf(acc[v] + bs, 0.01f);
    }
  }
}

// ---------------- embedding gathers --------------------------------------------------
__global__ void k_gather_circ(const int* __restrict__ cx, const float* __restrict__ emb,
                              float* __restrict__ xe) {
  int gid = blockIdx.x * blockDim.x + threadIdx.x;
  if (gid >= NC_ * 2 * F_) return;
  int n = gid / (2 * F_), k = gid - n * (2 * F_);
  int which = k / F_, c = k - which * F_;
  xe[gid] = emb[cx[n * 2 + which] * F_ + c];
}

__global__ void k_build_xt(const int* __restrict__ tx, const float* __restrict__ temb,
                           const float* __restrict__ xcirc, float* __restrict__ xt) {
  int gid = blockIdx.x * blockDim.x + threadIdx.x;
  if (gid >= NT_ * 3 * F_) return;
  int n = gid / (3 * F_), k = gid - n * (3 * F_);
  if (k < F_) xt[gid] = temb[tx[n] * F_ + k];
  else        xt[gid] = xcirc[((size_t)(n / Q_) * NGATE) * (2 * F_) + (k - F_)];
}

// ---------------- GATv2 edge pipeline ------------------------------------------------
__global__ void k_seg_init(float* __restrict__ mx, float* __restrict__ z,
                           float* __restrict__ aggr, int N, int H, int HC) {
  int gid = blockIdx.x * blockDim.x + threadIdx.x;
  if (gid < N * HC) aggr[gid] = 0.f;
  if (gid < N * H) { mx[gid] = -3.0e38f; z[gid] = 0.f; }
}

__global__ void k_edge_logit(const float* __restrict__ xl, const float* __restrict__ xr,
                             const int* __restrict__ ei, int E0, int Etot,
                             const float* __restrict__ att, float* __restrict__ lg,
                             float* __restrict__ mx, int H, int C) {
  int gid = blockIdx.x * blockDim.x + threadIdx.x;
  if (gid >= Etot * H) return;
  int e = gid / H, h = gid - e * H;
  int src = e < E0 ? ei[e] : e - E0;
  int dst = e < E0 ? ei[E0 + e] : e - E0;
  int HC = H * C;
  const float* pl = xl + (size_t)src * HC + h * C;
  const float* pr = xr + (size_t)dst * HC + h * C;
  const float* pa = att + h * C;
  float s = 0.f;
  for (int c = 0; c < C; ++c) {
    float t = pl[c] + pr[c];
    s += leakyf(t, 0.2f) * pa[c];
  }
  lg[gid] = s;
  atomicMaxF(&mx[dst * H + h], s);
}

__global__ void k_edge_soft(float* __restrict__ lg, const float* __restrict__ mx,
                            float* __restrict__ z, const int* __restrict__ ei,
                            int E0, int Etot, int H) {
  int gid = blockIdx.x * blockDim.x + threadIdx.x;
  if (gid >= Etot * H) return;
  int e = gid / H, h = gid - e * H;
  int dst = e < E0 ? ei[E0 + e] : e - E0;
  float w = __expf(lg[gid] - mx[dst * H + h]);
  lg[gid] = w;
  atomicAdd(&z[dst * H + h], w);
}

__global__ void k_edge_scatter(const float* __restrict__ xl, const float* __restrict__ lg,
                               const float* __restrict__ z, const int* __restrict__ ei,
                               int E0, int Etot, float* __restrict__ aggr, int H, int C) {
  int gid = blockIdx.x * blockDim.x + threadIdx.x;
  int HC = H * C;
  if (gid >= Etot * HC) return;
  int e = gid / HC, k = gid - e * HC;
  int h = k / C;
  int src = e < E0 ? ei[e] : e - E0;
  int dst = e < E0 ? ei[E0 + e] : e - E0;
  float alpha = lg[e * H + h] / (z[dst * H + h] + 1e-16f);
  atomicAdd(&aggr[(size_t)dst * HC + k], xl[(size_t)src * HC + k] * alpha);
}

__global__ void k_bias_act(const float* __restrict__ aggr, const float* __restrict__ bias,
                           const float* __restrict__ res, float* __restrict__ out,
                           int N, int C) {
  int gid = blockIdx.x * blockDim.x + threadIdx.x;
  if (gid >= N * C) return;
  int c = gid % C;
  float v = leakyf(aggr[gid] + bias[c], 0.01f);
  if (res) v += res[gid];
  out[gid] = v;
}

// ---------------- graph layer-norm ---------------------------------------------------
__global__ void k_ln_stats(const float* __restrict__ x, float* __restrict__ stats,
                           int P, int C) {
  int g = blockIdx.x;
  const float* p = x + (size_t)g * P * C;
  int n = P * C;
  float s = 0.f, s2 = 0.f;
  for (int i = threadIdx.x; i < n; i += blockDim.x) { float v = p[i]; s += v; s2 += v * v; }
  __shared__ float sh[512];
  sh[threadIdx.x] = s; sh[256 + threadIdx.x] = s2;
  __syncthreads();
  for (int o = 128; o > 0; o >>= 1) {
    if ((int)threadIdx.x < o) {
      sh[threadIdx.x] += sh[threadIdx.x + o];
      sh[256 + threadIdx.x] += sh[256 + threadIdx.x + o];
    }
    __syncthreads();
  }
  if (threadIdx.x == 0) {
    float mean = sh[0] / n;
    float var  = sh[256] / n - mean * mean;
    stats[g * 2] = mean;
    stats[g * 2 + 1] = rsqrtf(var + 1e-5f);
  }
}

__global__ void k_ln_apply(const float* __restrict__ x, const float* __restrict__ stats,
                           const float* __restrict__ w, const float* __restrict__ b,
                           const float* __restrict__ post, float* __restrict__ y,
                           int P, int C) {
  int gid = blockIdx.x * blockDim.x + threadIdx.x;
  if (gid >= B_ * P * C) return;
  int n = gid / C, c = gid - n * C;
  int g = n / P;
  float v = (x[gid] - stats[g * 2]) * stats[g * 2 + 1] * w[c] + b[c];
  if (post) v += post[gid];
  y[gid] = v;
}

// ---------------- per-pair tail: LN(128) -> W3 -> leaky -> W4 -------------------------
__global__ __launch_bounds__(256)
void k_pair_final(const float* __restrict__ h2, const float* __restrict__ lnw,
                  const float* __restrict__ lnb, const float* __restrict__ W3,
                  const float* __restrict__ b3, const float* __restrict__ W4,
                  const float* __restrict__ b4, float* __restrict__ o) {
  __shared__ float sh[8][128];
  int wid = threadIdx.x >> 5, lane = threadIdx.x & 31;
  int r = blockIdx.x * 8 + wid;
  const float* row = h2 + (size_t)r * 128;
  float v[4];
#pragma unroll
  for (int q = 0; q < 4; ++q) v[q] = row[lane + 32 * q];
  float s = v[0] + v[1] + v[2] + v[3];
#pragma unroll
  for (int off = 16; off; off >>= 1) s += __shfl_xor(s, off, 32);
  float mean = s * (1.f / 128.f);
  float s2 = 0.f;
#pragma unroll
  for (int q = 0; q < 4; ++q) { float d = v[q] - mean; s2 += d * d; }
#pragma unroll
  for (int off = 16; off; off >>= 1) s2 += __shfl_xor(s2, off, 32);
  float rstd = rsqrtf(s2 * (1.f / 128.f) + 1e-5f);
#pragma unroll
  for (int q = 0; q < 4; ++q) {
    int c = lane + 32 * q;
    sh[wid][c] = (v[q] - mean) * rstd * lnw[c] + lnb[c];
  }
  __syncthreads();
  float acc = b3[lane];
  for (int k = 0; k < 128; ++k) acc += sh[wid][k] * W3[k * 32 + lane];
  acc = leakyf(acc, 0.01f);
  float t = acc * W4[lane];
#pragma unroll
  for (int off = 16; off; off >>= 1) t += __shfl_xor(t, off, 32);
  if (lane == 0) o[r] = t + b4[0];
}

__global__ void k_sym(const float* __restrict__ ov, float* __restrict__ out) {
  int gid = blockIdx.x * blockDim.x + threadIdx.x;
  if (gid >= NPAIR) return;
  int b = gid >> 10, i = (gid >> 5) & 31, j = gid & 31;
  out[gid] = 0.5f * (ov[gid] + ov[(b << 10) + (j << 5) + i]);
}

// =====================================================================================
extern "C" void kernel_launch(void* const* d_in, const int* in_sizes, int n_in,
                              void* d_out, int out_size, void* d_ws, size_t ws_size,
                              hipStream_t stream) {
  (void)in_sizes; (void)n_in; (void)out_size; (void)ws_size;
  const int*   circ_x   = (const int*)d_in[0];
  const int*   circ_ei  = (const int*)d_in[1];
  const int*   topo_x   = (const int*)d_in[2];
  const int*   topo_ei  = (const int*)d_in[3];
  const float* circ_emb = (const float*)d_in[4];
  const float* circ_lnw = (const float*)d_in[35];
  const float* circ_lnb = (const float*)d_in[36];
  const float* topo_emb = (const float*)d_in[37];
  const float* topo_lnw = (const float*)d_in[68];
  const float* topo_lnb = (const float*)d_in[69];
  const float* W1  = (const float*)d_in[70];
  const float* b1  = (const float*)d_in[71];
  const float* W2  = (const float*)d_in[72];
  const float* b2  = (const float*)d_in[73];
  const float* mlw = (const float*)d_in[74];
  const float* mlb = (const float*)d_in[75];
  const float* W3  = (const float*)d_in[76];
  const float* b3  = (const float*)d_in[77];
  const float* W4  = (const float*)d_in[78];
  const float* b4  = (const float*)d_in[79];

  // -------- workspace layout (~202 MB) --------
  char* ws = (char*)d_ws;
  const size_t MB = 1024ull * 1024ull;
  _Float16* wt = (_Float16*)ws;                  // converted weights (<4MB)
  float* xe   = (float*)(ws + 4 * MB);           // 16MB  (circ xe / topo xt)
  float* xb0  = (float*)(ws + 20 * MB);          // 32MB  ping
  float* xb1  = (float*)(ws + 52 * MB);          // 32MB  pong
  float* aggr = (float*)(ws + 84 * MB);          // 32MB
  float* xl   = (float*)(ws + 116 * MB);         // 32MB  (also U)
  float* xr   = (float*)(ws + 148 * MB);         // 32MB  (also V)
  float* lg   = (float*)(ws + 180 * MB);         // 2MB edge logits/weights
  float* mx   = (float*)(ws + 182 * MB);         // segment max
  float* zz   = (float*)(ws + 183 * MB);         // segment sum
  float* ov   = (float*)(ws + 184 * MB);         // pair scalar out
  float* st   = (float*)(ws + 185 * MB);         // LN stats
  _Float16* xh = (_Float16*)(ws + 186 * MB);     // 16MB f16 activation staging
  float* h2   = xb1;                             // 64MB alias over xb1+aggr (adjacent)

  dim3 tb(256);
  auto blocks = [](long n) { return dim3((unsigned)((n + 255) / 256)); };

  // -------- convert all weights to f16 transposed --------
  size_t woff = 0;
  auto cvt = [&](const float* W, int Kin, int Co, int r0) -> _Float16* {
    _Float16* dstp = wt + woff;
    int tot = Kin * Co;
    k_convert<<<blocks(tot), tb, 0, stream>>>(W, dstp, Kin, Co, r0);
    woff += (size_t)tot;
    return dstp;
  };
  const int cCi[5] = {128, 256, 256, 256, 256};
  const int cCo[5] = {256, 256, 256, 256, 128};
  const int cH[5]  = {2, 2, 2, 2, 1};
  const int tCi[5] = {192, 384, 384, 384, 384};
  const int tCo[5] = {384, 384, 384, 384, 192};
  _Float16 *cwl[5], *cwr[5], *twl[5], *twr[5];
  for (int c = 0; c < 5; ++c) {
    cwl[c] = cvt((const float*)d_in[5 + c * 6 + 0], cCi[c], cCo[c], 0);
    cwr[c] = cvt((const float*)d_in[5 + c * 6 + 2], cCi[c], cCo[c], 0);
  }
  for (int c = 0; c < 5; ++c) {
    twl[c] = cvt((const float*)d_in[38 + c * 6 + 0], tCi[c], tCo[c], 0);
    twr[c] = cvt((const float*)d_in[38 + c * 6 + 2], tCi[c], tCo[c], 0);
  }
  _Float16* w1tT = cvt(W1, 192, 384, 0);     // top rows of W1
  _Float16* w1tB = cvt(W1, 192, 384, 192);   // bottom rows of W1
  _Float16* w2t  = cvt(W2, 384, 128, 0);

  // -------- one GATv2 conv stack --------
  auto run_stack = [&](int pbase, int N, int P, int E0, const int* ei, float* x0,
                       const int* Ci, const int* Co, const int* Hh,
                       _Float16* const* wlT, _Float16* const* wrT,
                       const float* lnw, const float* lnb) {
    int Etot = E0 + N;
    float* bufs[2] = {xb0, xb1};
    for (int c = 0; c < 5; ++c) {
      const float* bl   = (const float*)d_in[pbase + c * 6 + 1];
      const float* br   = (const float*)d_in[pbase + c * 6 + 3];
      const float* att  = (const float*)d_in[pbase + c * 6 + 4];
      const float* bias = (const float*)d_in[pbase + c * 6 + 5];
      const float* xin = (c == 0) ? x0 : bufs[(c + 1) & 1];
      float* xout = bufs[c & 1];
      int ci = Ci[c], co = Co[c], H = Hh[c], C = co / H;
      // convert layer input to f16 once; reused by both Wl and Wr GEMMs
      k_cvt_act<<<blocks((long)N * ci), tb, 0, stream>>>(xin, xh, N * ci);
      int waves = (N / 16) * (co / 64);
      dim3 gb((unsigned)((waves + 7) / 8));
      k_gemm<<<gb, tb, 0, stream>>>(xh, wlT[c], bl, xl, N, ci, co, 0);
      k_gemm<<<gb, tb, 0, stream>>>(xh, wrT[c], br, xr, N, ci, co, 0);
      k_seg_init<<<blocks((long)N * co), tb, 0, stream>>>(mx, zz, aggr, N, H, co);
      k_edge_logit<<<blocks((long)Etot * H), tb, 0, stream>>>(xl, xr, ei, E0, Etot, att, lg, mx, H, C);
      k_edge_soft<<<blocks((long)Etot * H), tb, 0, stream>>>(lg, mx, zz, ei, E0, Etot, H);
      k_edge_scatter<<<blocks((long)Etot * co), tb, 0, stream>>>(xl, lg, zz, ei, E0, Etot, aggr, H, C);
      const float* res = (c == 4) ? x0 : nullptr;
      float* dst = (c == 4) ? xb0 : xout;
      k_bias_act<<<blocks((long)N * co), tb, 0, stream>>>(aggr, bias, res, dst, N, co);
    }
    int cf = Co[4];
    k_ln_stats<<<dim3(B_), tb, 0, stream>>>(xb0, st, P, cf);
    k_ln_apply<<<blocks((long)N * cf), tb, 0, stream>>>(xb0, st, lnw, lnb, x0, xb0, P, cf);
  };

  // -------- circuit GNN --------
  k_gather_circ<<<blocks((long)NC_ * 2 * F_), tb, 0, stream>>>(circ_x, circ_emb, xe);
  run_stack(5, NC_, NGATE, EC_, circ_ei, xe, cCi, cCo, cH, cwl, cwr, circ_lnw, circ_lnb);
  // circuit result (with +xe residual) in xb0 [NC,128]; circ_feat = row b*NGATE

  // -------- topo GNN --------
  k_build_xt<<<blocks((long)NT_ * 3 * F_), tb, 0, stream>>>(topo_x, topo_emb, xb0, xe);
  run_stack(38, NT_, Q_, ET_, topo_ei, xe, tCi, tCo, cH, twl, twr, topo_lnw, topo_lnb);
  // xg in xb0 [NT,192]

  // -------- pair MLP (factorized layer 1, fused layer 2) --------
  {
    k_cvt_act<<<blocks((long)NT_ * 192), tb, 0, stream>>>(xb0, xh, NT_ * 192);
    int waves = (NT_ / 16) * (384 / 64);
    dim3 gb((unsigned)((waves + 7) / 8));
    k_gemm<<<gb, tb, 0, stream>>>(xh, w1tT, nullptr, xl, NT_, 192, 384, 0);  // U
    k_gemm<<<gb, tb, 0, stream>>>(xh, w1tB, b1,      xr, NT_, 192, 384, 0);  // V + b1
  }
  k_gemm_pair<<<dim3(((NPAIR / 16) * 2 + 7) / 8), tb, 0, stream>>>(xl, xr, w2t, b2, h2);
  k_pair_final<<<dim3(NPAIR / 8), tb, 0, stream>>>(h2, mlw, mlb, W3, b3, W4, b4, ov);
  k_sym<<<blocks(NPAIR), tb, 0, stream>>>(ov, (float*)d_out);
}
